// Attention_5918464934573
// MI455X (gfx1250) — matrix operand order repaired and checked by reading, not compiled, not run on hardware
//
#include <hip/hip_runtime.h>
#include <hip/hip_bf16.h>

// Problem constants (match reference)
#define BATCH    2
#define SEQ      2048
#define DIM      1024
#define HEADS    16
#define KV_HEADS 4
#define HEAD_DIM 64
#define KVDIM    256          // KV_HEADS * HEAD_DIM
#define SCALE    0.125f       // 1/sqrt(64)

typedef __bf16 bf16_t;
typedef __bf16 v16bf __attribute__((ext_vector_type(16)));
typedef __bf16 v8bf  __attribute__((ext_vector_type(8)));
typedef float  v8f   __attribute__((ext_vector_type(8)));
typedef int    v4i   __attribute__((ext_vector_type(4)));

#define AS1 __attribute__((address_space(1)))
#define AS3 __attribute__((address_space(3)))

#if __has_builtin(__builtin_amdgcn_global_load_async_to_lds_b128)
#define HAVE_ASYNC_LDS 1
#else
#define HAVE_ASYNC_LDS 0
#endif

#define WMMA_BF16(a,b,c) \
  __builtin_amdgcn_wmma_f32_16x16x32_bf16(false,(a),false,(b),(short)0,(c),false,false)

static __device__ __forceinline__ bf16_t f2bf(float f) {
    union { float f; unsigned u; } uf; uf.f = f;
    unsigned u = uf.u + 0x7FFFu + ((uf.u >> 16) & 1u);   // round-to-nearest-even
    union { unsigned short s; bf16_t b; } ub; ub.s = (unsigned short)(u >> 16);
    return ub.b;
}

// A-fragment 16(M)x32(K), bf16, from row-major [row][ld] storage.
// ISA layout: lanes 0-15 hold row M=lane, K=0..7 & 16..23; lanes 16-31 row M=lane-16, K=8..15 & 24..31.
static __device__ __forceinline__ v16bf load_a16x32(const bf16_t* base, int ld, int row0, int col0) {
    const int lane = threadIdx.x & 31, grp = lane >> 4, r = lane & 15;
    const bf16_t* p = base + (size_t)(row0 + r) * ld + col0 + grp * 8;
    union { v16bf v; v8bf h[2]; } u;
    u.h[0] = *(const v8bf*)p;          // K = grp*8 .. grp*8+7
    u.h[1] = *(const v8bf*)(p + 16);   // K = 16+grp*8 .. 16+grp*8+7
    return u.v;
}

// B-fragment 32(K)x16(N), bf16, from "B-transposed" storage: row index = N, col = K.
// ISA layout: lanes 0-15 hold N=lane, K=0..7 & 16..23; lanes 16-31 hold N=lane-16, K=8..15 & 24..31.
static __device__ __forceinline__ v16bf load_b32x16(const bf16_t* baseT, int ld, int n0, int k0) {
    const int lane = threadIdx.x & 31, grp = lane >> 4, r = lane & 15;
    const bf16_t* p = baseT + (size_t)(n0 + r) * ld + k0 + grp * 8;
    union { v16bf v; v8bf h[2]; } u;
    u.h[0] = *(const v8bf*)p;
    u.h[1] = *(const v8bf*)(p + 16);
    return u.v;
}

// Stage 16 bytes of B tile into LDS (async DMA on gfx1250 if the builtin exists).
static __device__ __forceinline__ void stage16(const bf16_t* gsrc, bf16_t* ldst) {
#if HAVE_ASYNC_LDS
    // Builtin signature: (v4i AS1* gsrc, v4i AS3* lds, imm offset, imm cpol).
    // Generic LDS pointer's low 32 bits are the LDS offset.
    __builtin_amdgcn_global_load_async_to_lds_b128(
        (AS1 v4i*)(unsigned long long)(size_t)gsrc,
        (AS3 v4i*)(unsigned int)(size_t)ldst, 0, 0);
#else
    *(v8bf*)ldst = *(const v8bf*)gsrc;
#endif
}

// ---------------- conversion kernels ----------------
__global__ void cvt_copy_kernel(const float* __restrict__ src, bf16_t* __restrict__ dst, int n) {
    int i = blockIdx.x * blockDim.x + threadIdx.x;
    if (i < n) dst[i] = f2bf(src[i]);
}

__global__ void cvt_transpose_kernel(const float* __restrict__ src, bf16_t* __restrict__ dst,
                                     int K, int N) {
    int i = blockIdx.x * blockDim.x + threadIdx.x;
    if (i < K * N) {
        int k = i / N, n = i % N;
        dst[(size_t)n * K + k] = f2bf(src[i]);
    }
}

// ---------------- GEMM: C[M,N] = A[M,K] * Bt[N,K]^T + bias ----------------
// block = 128 threads = 4 waves; wave computes 64(M)x32(N); block tile 256(M) x 32(N).
// B tile (32N x 32K bf16 = 2 KB) is double-buffered in LDS, staged asynchronously and
// shared by all 4 waves; A fragments stream straight from global (L2-resident).
// mode 0: bf16 row-major out; mode 1: bf16 V^T out ([b][n][tok]); mode 2: f32 row-major out.
__global__ void gemm_bf16_kernel(const bf16_t* __restrict__ A, const bf16_t* __restrict__ Bt,
                                 const float* __restrict__ bias, void* outp,
                                 int M, int N, int K, int mode) {
    __shared__ bf16_t Bs[2][32][32];
    const int tid  = threadIdx.x;
    const int w    = tid >> 5;
    const int lane = tid & 31, grp = lane >> 4, r = lane & 15;
    const int m0   = blockIdx.x * 256 + w * 64;
    const int n0   = blockIdx.y * 32;

    // B-tile staging assignment: each of 128 threads moves 16 bytes.
    const int srow = tid >> 2;          // 0..31 : N index within tile
    const int scol = (tid & 3) * 8;     // half offset within row (0,8,16,24)
    const bf16_t* gB = Bt + (size_t)(n0 + srow) * K + scol;
    bf16_t* lB0 = &Bs[0][srow][scol];
    bf16_t* lB1 = &Bs[1][srow][scol];

    v8f zero = {};
    v8f acc[4][2];
    #pragma unroll
    for (int i = 0; i < 4; i++) { acc[i][0] = zero; acc[i][1] = zero; }

    stage16(gB, lB0);                               // prologue: tile for k = 0

    for (int k = 0, it = 0; k < K; k += 32, ++it) {
        const int cur = it & 1;
        const bool more = (k + 32) < K;
        if (more) stage16(gB + (k + 32), cur ? lB0 : lB1);   // prefetch next tile
#if HAVE_ASYNC_LDS
        if (more) asm volatile("s_wait_asynccnt 0x1" ::: "memory");
        else      asm volatile("s_wait_asynccnt 0x0" ::: "memory");
#endif
        __syncthreads();                            // current B tile visible to all waves

        const bf16_t* Bl = &Bs[cur][0][0];
        v16bf b0 = load_b32x16(Bl, 32, 0,  0);
        v16bf b1 = load_b32x16(Bl, 32, 16, 0);

        __builtin_prefetch(A + (size_t)(m0 + r) * K + k + 64, 0, 3);
        v16bf a0 = load_a16x32(A, K, m0,      k);
        v16bf a1 = load_a16x32(A, K, m0 + 16, k);
        v16bf a2 = load_a16x32(A, K, m0 + 32, k);
        v16bf a3 = load_a16x32(A, K, m0 + 48, k);

        acc[0][0] = WMMA_BF16(a0, b0, acc[0][0]);
        acc[0][1] = WMMA_BF16(a0, b1, acc[0][1]);
        acc[1][0] = WMMA_BF16(a1, b0, acc[1][0]);
        acc[1][1] = WMMA_BF16(a1, b1, acc[1][1]);
        acc[2][0] = WMMA_BF16(a2, b0, acc[2][0]);
        acc[2][1] = WMMA_BF16(a2, b1, acc[2][1]);
        acc[3][0] = WMMA_BF16(a3, b0, acc[3][0]);
        acc[3][1] = WMMA_BF16(a3, b1, acc[3][1]);

        __syncthreads();                            // reads done before buffer reuse
    }

    #pragma unroll
    for (int i = 0; i < 4; i++) {
        #pragma unroll
        for (int j = 0; j < 2; j++) {
            v8f c = acc[i][j];
            int col = n0 + j * 16 + r;
            float bval = bias[col];
            #pragma unroll
            for (int vr = 0; vr < 8; vr++) {
                int row   = m0 + i * 16 + vr + 8 * grp;
                float val = c[vr] + bval;
                if (mode == 0) {
                    ((bf16_t*)outp)[(size_t)row * N + col] = f2bf(val);
                } else if (mode == 1) {
                    int b = row >> 11, tok = row & (SEQ - 1);
                    ((bf16_t*)outp)[((size_t)b * KVDIM + col) * SEQ + tok] = f2bf(val);
                } else {
                    ((float*)outp)[(size_t)row * N + col] = val;
                }
            }
        }
    }
}

// ---------------- Flash attention ----------------
// grid: (SEQ/64, BATCH*HEADS), block 128 (4 waves). Wave handles 16 query rows.
__global__ void attn_kernel(const bf16_t* __restrict__ Qb, const bf16_t* __restrict__ Kb,
                            const bf16_t* __restrict__ Vt, bf16_t* __restrict__ ctx) {
    __shared__ float plds[4][16 * 32];
    const int w    = threadIdx.x >> 5;
    const int lane = threadIdx.x & 31, grp = lane >> 4, r = lane & 15;
    const int bh   = blockIdx.y;
    const int b = bh / HEADS, h = bh % HEADS, kvh = h >> 2;
    const int q0   = blockIdx.x * 64 + w * 16;
    const int qrow = b * SEQ + q0;

    v16bf aq0 = load_a16x32(Qb, DIM, qrow, h * HEAD_DIM);
    v16bf aq1 = load_a16x32(Qb, DIM, qrow, h * HEAD_DIM + 32);

    float mi[8], li[8];
    #pragma unroll
    for (int t = 0; t < 8; t++) { mi[t] = -1e30f; li[t] = 0.0f; }
    v8f o0 = {}, o1 = {}, o2 = {}, o3 = {};

    const bf16_t* VtB = Vt + (size_t)b * KVDIM * SEQ;
    float* pl = plds[w];

    for (int kb = 0; kb < SEQ; kb += 32) {
        // S = Q * K^T  (two 16x16 key sub-tiles, d in two 32-chunks)
        v8f s0 = {}, s1 = {};
        v16bf bk;
        bk = load_b32x16(Kb, KVDIM, b * SEQ + kb,      kvh * HEAD_DIM);      s0 = WMMA_BF16(aq0, bk, s0);
        bk = load_b32x16(Kb, KVDIM, b * SEQ + kb,      kvh * HEAD_DIM + 32); s0 = WMMA_BF16(aq1, bk, s0);
        bk = load_b32x16(Kb, KVDIM, b * SEQ + kb + 16, kvh * HEAD_DIM);      s1 = WMMA_BF16(aq0, bk, s1);
        bk = load_b32x16(Kb, KVDIM, b * SEQ + kb + 16, kvh * HEAD_DIM + 32); s1 = WMMA_BF16(aq1, bk, s1);

        // Online softmax update (row = vr + 8*grp; columns striped across 16-lane group)
        #pragma unroll
        for (int t = 0; t < 8; t++) {
            float a = s0[t] * SCALE, c = s1[t] * SCALE;
            float mx = fmaxf(a, c);
            #pragma unroll
            for (int off = 8; off > 0; off >>= 1) mx = fmaxf(mx, __shfl_xor(mx, off, 32));
            float mn    = fmaxf(mi[t], mx);
            float alpha = __expf(mi[t] - mn);
            mi[t] = mn;
            float p0 = __expf(a - mn), p1 = __expf(c - mn);
            s0[t] = p0; s1[t] = p1;
            float rs = p0 + p1;
            #pragma unroll
            for (int off = 8; off > 0; off >>= 1) rs += __shfl_xor(rs, off, 32);
            li[t] = li[t] * alpha + rs;
            o0[t] *= alpha; o1[t] *= alpha; o2[t] *= alpha; o3[t] *= alpha;
        }

        // Repack P (f32 C-layout 16x32) -> bf16 A-layout via per-wave LDS
        #pragma unroll
        for (int t = 0; t < 8; t++) {
            int row = t + 8 * grp;
            pl[row * 32 + r]      = s0[t];
            pl[row * 32 + 16 + r] = s1[t];
        }
        __builtin_amdgcn_wave_barrier();
        asm volatile("s_wait_dscnt 0" ::: "memory");
        union { v16bf v; bf16_t e[16]; } up;
        const float* prow = &pl[r * 32 + grp * 8];
        #pragma unroll
        for (int e = 0; e < 8; e++) up.e[e]     = f2bf(prow[e]);
        #pragma unroll
        for (int e = 0; e < 8; e++) up.e[8 + e] = f2bf(prow[16 + e]);
        __builtin_amdgcn_wave_barrier();
        v16bf pa = up.v;

        // O += P * V  (V^T storage: row = d, col = token => contiguous B fragments)
        o0 = WMMA_BF16(pa, load_b32x16(VtB, SEQ, kvh * HEAD_DIM +  0, kb), o0);
        o1 = WMMA_BF16(pa, load_b32x16(VtB, SEQ, kvh * HEAD_DIM + 16, kb), o1);
        o2 = WMMA_BF16(pa, load_b32x16(VtB, SEQ, kvh * HEAD_DIM + 32, kb), o2);
        o3 = WMMA_BF16(pa, load_b32x16(VtB, SEQ, kvh * HEAD_DIM + 48, kb), o3);
    }

    // Epilogue: O /= l, store bf16 ctx [tok][h*64+d]
    #pragma unroll
    for (int t = 0; t < 8; t++) {
        float inv = 1.0f / li[t];
        int row = qrow + t + 8 * grp;
        size_t base = (size_t)row * DIM + h * HEAD_DIM + r;
        ctx[base +  0] = f2bf(o0[t] * inv);
        ctx[base + 16] = f2bf(o1[t] * inv);
        ctx[base + 32] = f2bf(o2[t] * inv);
        ctx[base + 48] = f2bf(o3[t] * inv);
    }
}

// ---------------- launch ----------------
extern "C" void kernel_launch(void* const* d_in, const int* in_sizes, int n_in,
                              void* d_out, int out_size, void* d_ws, size_t ws_size,
                              hipStream_t stream) {
    const float* x  = (const float*)d_in[0];
    const float* Wq = (const float*)d_in[1];
    const float* bq = (const float*)d_in[2];
    const float* Wk = (const float*)d_in[3];
    const float* bk = (const float*)d_in[4];
    const float* Wv = (const float*)d_in[5];
    const float* bv = (const float*)d_in[6];
    const float* Wo = (const float*)d_in[7];
    const float* bo = (const float*)d_in[8];

    bf16_t* ws = (bf16_t*)d_ws;
    size_t off = 0;
    bf16_t* xb  = ws + off; off += (size_t)4096 * 1024;   // x bf16
    bf16_t* WqT = ws + off; off += (size_t)1024 * 1024;   // Wq^T bf16
    bf16_t* WkT = ws + off; off += (size_t)1024 * 256;
    bf16_t* WvT = ws + off; off += (size_t)1024 * 256;
    bf16_t* WoT = ws + off; off += (size_t)1024 * 1024;
    bf16_t* Qb  = ws + off; off += (size_t)4096 * 1024;   // Q bf16 [tok][h*64+d]
    bf16_t* Kb  = ws + off; off += (size_t)4096 * 256;    // K bf16 [tok][kvh*64+d]
    bf16_t* Vt  = ws + off; off += (size_t)4096 * 256;    // V^T bf16 [b][kvh*64+d][tok]
    bf16_t* Ctx = ws + off; off += (size_t)4096 * 1024;   // attention output bf16

    cvt_copy_kernel<<<(4096 * 1024) / 256, 256, 0, stream>>>(x, xb, 4096 * 1024);
    cvt_transpose_kernel<<<(1024 * 1024) / 256, 256, 0, stream>>>(Wq, WqT, 1024, 1024);
    cvt_transpose_kernel<<<(1024 *  256) / 256, 256, 0, stream>>>(Wk, WkT, 1024, 256);
    cvt_transpose_kernel<<<(1024 *  256) / 256, 256, 0, stream>>>(Wv, WvT, 1024, 256);
    cvt_transpose_kernel<<<(1024 * 1024) / 256, 256, 0, stream>>>(Wo, WoT, 1024, 1024);

    gemm_bf16_kernel<<<dim3(16, 32), 128, 0, stream>>>(xb,  WqT, bq, Qb,    4096, 1024, 1024, 0);
    gemm_bf16_kernel<<<dim3(16,  8), 128, 0, stream>>>(xb,  WkT, bk, Kb,    4096,  256, 1024, 0);
    gemm_bf16_kernel<<<dim3(16,  8), 128, 0, stream>>>(xb,  WvT, bv, Vt,    4096,  256, 1024, 1);
    attn_kernel<<<dim3(32, 32), 128, 0, stream>>>(Qb, Kb, Vt, Ctx);
    gemm_bf16_kernel<<<dim3(16, 32), 128, 0, stream>>>(Ctx, WoT, bo, d_out, 4096, 1024, 1024, 2);
}
